// MultiHeadAttentionRoPE_25357486915975
// MI455X (gfx1250) — matrix operand order repaired and checked by reading, not compiled, not run on hardware
//
#include <hip/hip_runtime.h>
#include <hip/hip_bf16.h>

// ---------------------------------------------------------------------------
// MI455X (gfx1250) multi-head attention + RoPE, f16 WMMA path, fp32 accumulate
// TDM (tensor_load_to_lds) stages V chunks into LDS, overlapped with S/softmax
// ---------------------------------------------------------------------------

typedef __attribute__((ext_vector_type(16))) _Float16 v16h;
typedef __attribute__((ext_vector_type(8)))  _Float16 v8h;
typedef __attribute__((ext_vector_type(8)))  float    v8f;
typedef unsigned int u32x4 __attribute__((ext_vector_type(4)));
typedef int i32x4 __attribute__((ext_vector_type(4)));
typedef int i32x8 __attribute__((ext_vector_type(8)));

#define NB    2
#define NT    2048
#define NC    1024
#define NH    16
#define HD    64
#define NM    (NB * NT)        // 4096 rows in token-major GEMMs
#define N3C   (3 * NC)         // 3072
#define NEGINF (-3.0e38f)

#define CHUNK  256
#define SPITCH 257             // odd pitch -> conflict-free strided LDS reads
#define VPITCH (CHUNK + 8)     // 264 halves: matches TDM pad (4 dw / 128 dw)

union AFragU { v16h v; v8h h[2]; };

static __device__ inline v8f vzero8() {
  v8f z = {0.f, 0.f, 0.f, 0.f, 0.f, 0.f, 0.f, 0.f};
  return z;
}

// 16-bit A matrix 16x32 (MxK), row-major source with leading dim lda (halves).
// ISA 7.12.2: lanes 0-15 M=lane, lanes 16-31 M=lane-16;
// VGPR0-3: K=0..7 (lo lanes) / K=8..15 (hi lanes); VGPR4-7: +16.
static __device__ inline v16h load_a_frag(const _Float16* __restrict__ base,
                                          int lda, int m0, int k0) {
  const int lane = threadIdx.x & 31;
  const int row  = m0 + (lane & 15);
  const int hi   = lane >> 4;
  const _Float16* p = base + (long)row * lda + k0 + hi * 8;
  AFragU f;
  f.h[0] = *(const v8h*)(p);        // K = k0 + hi*8 + [0..7]
  f.h[1] = *(const v8h*)(p + 16);   // K = k0 + 16 + hi*8 + [0..7]
  return f.v;
}

// 16-bit B matrix 32x16 (KxN) read from row-major B^T (N rows, K contiguous),
// leading dim ldb (halves). N=lane%16; elems 0-7: K=k0+hi*8+[0..7];
// elems 8-15: K=k0+16+hi*8+[0..7] (same K order as load_a_frag).
static __device__ inline v16h load_b_frag(const _Float16* __restrict__ base,
                                          int ldb, int n0, int k0) {
  const int lane = threadIdx.x & 31;
  const int col  = n0 + (lane & 15);
  const int hi   = lane >> 4;
  const _Float16* p = base + (long)col * ldb + k0 + hi * 8;
  AFragU f;
  f.h[0] = *(const v8h*)(p);
  f.h[1] = *(const v8h*)(p + 16);
  return f.v;
}

static __device__ inline v8f wmma16(v16h a, v16h b, v8f c) {
  // v_wmma_f32_16x16x32_f16, fp32 accumulate
  return __builtin_amdgcn_wmma_f32_16x16x32_f16(false, a, false, b,
                                                (short)0, c, false, false);
}

// ---------------------------------------------------------------------------
// TDM: DMA a 64-row x 256-half tile of V^T (row stride NT halves) into LDS,
// inserting a 4-dword pad every 128 dwords (-> VPITCH-half LDS rows).
static __device__ inline void tdm_stage_v(const _Float16* gsrc, void* ldsdst) {
#if defined(__AMDGCN__) && __has_builtin(__builtin_amdgcn_tensor_load_to_lds)
  const unsigned long long gaddr = (unsigned long long)(__SIZE_TYPE__)gsrc;
  const unsigned ldsoff = (unsigned)(unsigned long long)(__SIZE_TYPE__)ldsdst;
  u32x4 g0;
  g0[0] = 1u;                                   // count=1 (valid descriptor)
  g0[1] = ldsoff;                               // lds_addr (bytes)
  g0[2] = (unsigned)(gaddr & 0xffffffffu);      // global_addr[31:0]
  g0[3] = (unsigned)((gaddr >> 32) & 0x01ffffffu) | (2u << 30);  // type=2
  i32x8 g1;
  g1[0] = (1 << 16)        // data_size = 2 bytes
        | (1 << 20)        // pad_enable
        | (6 << 22)        // pad_interval: 2^(6+1) = 128 dwords
        | (3 << 25);       // pad_amount: 3+1 = 4 dwords
  g1[1] = (int)((unsigned)(NT & 0xffff) << 16);          // tensor_dim0 lo16
  g1[2] = (NT >> 16) | (HD << 16);                       // dim0 hi | dim1 lo
  g1[3] = (0) | (CHUNK << 16);                           // dim1 hi | tile_dim0
  g1[4] = HD;                                            // tile_dim1 (rows)
  g1[5] = NT;                                            // dim0_stride lo32
  g1[6] = 0;
  g1[7] = 0;
  i32x4 gz = {0, 0, 0, 0};
#if __clang_major__ >= 23
  i32x8 gz8 = {0, 0, 0, 0, 0, 0, 0, 0};
  __builtin_amdgcn_tensor_load_to_lds(g0, g1, gz, gz, gz8, 0);
#else
  __builtin_amdgcn_tensor_load_to_lds(g0, g1, gz, gz, 0);
#endif
#else
  // Fallback: this wave copies the tile with plain loads/stores.
  const int lane = threadIdx.x & 31;
  _Float16(*dst)[VPITCH] = (_Float16(*)[VPITCH])ldsdst;
  for (int r = 0; r < HD; ++r) {
    v8h* drow = (v8h*)dst[r];
    const v8h* srow = (const v8h*)(gsrc + (long)r * NT);
    for (int cseg = lane; cseg < CHUNK / 8; cseg += 32) drow[cseg] = srow[cseg];
  }
#endif
}

static __device__ inline void tdm_wait() {
#if defined(__AMDGCN__) && __has_builtin(__builtin_amdgcn_s_wait_tensorcnt)
  __builtin_amdgcn_s_wait_tensorcnt(0);
#endif
}

// ---------------------------------------------------------------------------
__global__ __launch_bounds__(256) void cvt_kernel(const float* __restrict__ src,
                                                  _Float16* __restrict__ dst,
                                                  int n) {
  int i = blockIdx.x * 256 + threadIdx.x;
  if (i < n) dst[i] = (_Float16)src[i];
}

// ---------------------------------------------------------------------------
// Scatter one 16x16 C tile of the QKV GEMM with RoPE (q/k) or transpose (v).
static __device__ inline void scatter_qkv(v8f acc, int m0, int n0,
                                          _Float16* __restrict__ qf,
                                          _Float16* __restrict__ kf,
                                          _Float16* __restrict__ vt) {
  const int lane = threadIdx.x & 31;
  const int hi = lane >> 4;
  const int n = n0 + (lane & 15);
  const int sec = n >> 10;     // 0:q 1:k 2:v (uniform across the wave)
  const int sn = n & (NC - 1);
  const int h = sn >> 6, d = sn & 63;

  if (sec < 2) {
    // RoPE: inv_freq = 10000^{-(d/2)/32}
    const float inv_freq = __expf(-(float)(d >> 1) * 0.28782313662425572f);
    _Float16* __restrict__ dst = (sec == 0) ? qf : kf;
#pragma unroll
    for (int r = 0; r < 8; ++r) {
      int m = m0 + r + hi * 8;
      int bb = m >> 11;          // /NT
      int t  = m & (NT - 1);
      float v  = acc[r];
      float pv = __shfl_xor(v, 1, 32);   // partner: d^1 (adjacent lane)
      float ang = (float)t * inv_freq;
      float sn_, cs_;
      __sincosf(ang, &sn_, &cs_);
      float o = (d & 1) ? (v * cs_ + pv * sn_) : (v * cs_ - pv * sn_);
      dst[(((long)(bb * NH + h) * NT + t) << 6) + d] = (_Float16)o;
    }
  } else {
#pragma unroll
    for (int r = 0; r < 8; ++r) {
      int m = m0 + r + hi * 8;
      int bb = m >> 11;
      int t  = m & (NT - 1);
      // V stored transposed: [b][h][d][t]
      vt[((long)(bb * NH + h) * HD + d) * NT + t] = (_Float16)acc[r];
    }
  }
}

// QKV projection (4096x3072 = x[4096x1024] @ wqkv^T) + RoPE + scatter.
// One wave per 32x32 output tile (2x2 WMMA register tiling); 8 waves / block.
__global__ __launch_bounds__(256) void qkv_rope_kernel(
    const _Float16* __restrict__ xh, const _Float16* __restrict__ wqh,
    _Float16* __restrict__ qf, _Float16* __restrict__ kf,
    _Float16* __restrict__ vt) {
  const int waveId = blockIdx.x * 8 + (threadIdx.x >> 5);
  const int nT = waveId % (N3C / 32);   // 96
  const int mT = waveId / (N3C / 32);   // 128
  const int m0 = mT * 32, n0 = nT * 32;

  v8f c00 = vzero8(), c01 = vzero8(), c10 = vzero8(), c11 = vzero8();
#pragma unroll 2
  for (int k0 = 0; k0 < NC; k0 += 32) {
    v16h a0 = load_a_frag(xh, NC, m0, k0);
    v16h a1 = load_a_frag(xh, NC, m0 + 16, k0);
    v16h b0 = load_b_frag(wqh, NC, n0, k0);
    v16h b1 = load_b_frag(wqh, NC, n0 + 16, k0);
    c00 = wmma16(a0, b0, c00);
    c01 = wmma16(a0, b1, c01);
    c10 = wmma16(a1, b0, c10);
    c11 = wmma16(a1, b1, c11);
  }
  scatter_qkv(c00, m0,      n0,      qf, kf, vt);
  scatter_qkv(c01, m0,      n0 + 16, qf, kf, vt);
  scatter_qkv(c10, m0 + 16, n0,      qf, kf, vt);
  scatter_qkv(c11, m0 + 16, n0 + 16, qf, kf, vt);
}

// ---------------------------------------------------------------------------
// Flash attention: one block per (b,h, 16 query rows); online softmax over
// 256-column chunks staged in LDS. All 8 waves build S; all 8 waves run P@V
// with the chunk's K-range split between wave pairs (w, w+4). V chunks are
// DMA'd into LDS by the Tensor Data Mover, overlapped with S/softmax.
__global__ __launch_bounds__(256) void attn_kernel(
    const _Float16* __restrict__ qf, const _Float16* __restrict__ kf,
    const _Float16* __restrict__ vt, _Float16* __restrict__ attn) {
  __shared__ float S[16][SPITCH];
  __shared__ _Float16 VF[HD][VPITCH];       // TDM destination (padded rows)
  __shared__ float rowM[16], rowL[16], rowA[16];

  const int bh  = blockIdx.x >> 7;     // (b*NH + h)
  const int qt  = blockIdx.x & 127;
  const int tq0 = qt * 16;
  const _Float16* __restrict__ qp = qf + (long)bh * NT * HD;
  const _Float16* __restrict__ kp = kf + (long)bh * NT * HD;
  const _Float16* __restrict__ vp = vt + (long)bh * HD * NT;

  const int wave = threadIdx.x >> 5;
  const int lane = threadIdx.x & 31;
  const int hi = lane >> 4, lo = lane & 15;
  const int dcol  = (wave & 3) * 16;   // which 16 of the 64 d-columns
  const int khalf = wave >> 2;         // which half of the chunk's K range

  // Q fragments (K = d, two 32-wide steps)
  v16h qa0 = load_a_frag(qp, HD, tq0, 0);
  v16h qa1 = load_a_frag(qp, HD, tq0, 32);

  if (threadIdx.x < 16) { rowM[threadIdx.x] = NEGINF; rowL[threadIdx.x] = 0.f; }
  v8f o = vzero8();

  const int nChunks = tq0 / CHUNK + 1;
  for (int j = 0; j < nChunks; ++j) {
    const int cb = j * CHUNK;

    // ---- kick off V-chunk DMA (completes during S/softmax phases) --------
    if (wave == 0) tdm_stage_v(vp + cb, (void*)&VF[0][0]);

    // ---- S chunk: 16 k-tiles, wave does tiles {wave, wave+8} -------------
    for (int tt = wave; tt < 16; tt += 8) {
      const int tk0 = cb + tt * 16;
      v8f c = vzero8();
      if (tk0 <= tq0 + 15) {
        v16h b0 = load_b_frag(kp, HD, tk0, 0);
        v16h b1 = load_b_frag(kp, HD, tk0, 32);
        c = wmma16(qa0, b0, c);
        c = wmma16(qa1, b1, c);
      }
      const int tk = tk0 + lo;
#pragma unroll
      for (int r = 0; r < 8; ++r) {
        int m = r + hi * 8;
        float sv = (tk <= tq0 + m) ? c[r] * 0.125f : NEGINF;  // hd^-0.5 = 1/8
        S[m][tt * 16 + lo] = sv;
      }
    }
    __syncthreads();

    // ---- online softmax update: 16 threads per row -----------------------
    {
      const int row = threadIdx.x >> 4;
      const int cix = threadIdx.x & 15;
      float* srow = S[row];
      float mloc = NEGINF;
      for (int cc = cix; cc < CHUNK; cc += 16) mloc = fmaxf(mloc, srow[cc]);
#pragma unroll
      for (int off = 8; off; off >>= 1)
        mloc = fmaxf(mloc, __shfl_xor(mloc, off, 16));
      float mold = rowM[row];
      float mnew = fmaxf(mold, mloc);
      float alpha = __expf(mold - mnew);
      float sloc = 0.f;
      for (int cc = cix; cc < CHUNK; cc += 16) {
        float p = __expf(srow[cc] - mnew);
        srow[cc] = p;
        sloc += p;
      }
#pragma unroll
      for (int off = 8; off; off >>= 1) sloc += __shfl_xor(sloc, off, 16);
      if (cix == 0) {
        rowA[row] = alpha;
        rowM[row] = mnew;
        rowL[row] = rowL[row] * alpha + sloc;
      }
    }
    // Wave 0 fences the TDM; the barrier publishes completion to all waves.
    if (wave == 0) tdm_wait();
    __syncthreads();

    // ---- O += P @ V : all 8 waves; wave pair (w, w+4) splits the K range -
#pragma unroll
    for (int r = 0; r < 8; ++r) o[r] *= rowA[r + hi * 8];
#pragma unroll 2
    for (int kq = 0; kq < 4; ++kq) {
      const int kt = khalf * 4 + kq;
      // A fragment (P, f32 in LDS -> f16), row = lo, K pattern per ISA
      const float* pr = S[lo];
      const int kb = kt * 32 + hi * 8;
      v16h a;
#pragma unroll
      for (int e = 0; e < 8; ++e) a[e] = (_Float16)pr[kb + e];
#pragma unroll
      for (int e = 0; e < 8; ++e) a[8 + e] = (_Float16)pr[kb + 16 + e];
      // B fragment (V) from the TDM-staged LDS tile (padded row pitch)
      const _Float16* vls = &VF[dcol + lo][kt * 32 + hi * 8];
      AFragU bf;
      bf.h[0] = *(const v8h*)(vls);
      bf.h[1] = *(const v8h*)(vls + 16);
      o = wmma16(a, bf.v, o);
    }
    __syncthreads();
  }

  // Combine the two K-half partials through LDS (S is free now), then store.
  float* OSf = &S[0][0];                // 4*16*16 floats, fits easily
  if (khalf == 1) {
#pragma unroll
    for (int r = 0; r < 8; ++r)
      OSf[(wave & 3) * 256 + (r + hi * 8) * 16 + lo] = o[r];
  }
  __syncthreads();
  if (khalf == 0) {
    const int bb = bh >> 4, h = bh & 15;
#pragma unroll
    for (int r = 0; r < 8; ++r) {
      int m = r + hi * 8;
      float val = (o[r] + OSf[(wave & 3) * 256 + m * 16 + lo]) / rowL[m];
      attn[((long)(bb * NT + tq0 + m) << 10) + h * 64 + dcol + lo] =
          (_Float16)val;
    }
  }
}

// ---------------------------------------------------------------------------
// Output projection: out[4096x1024] = attn @ w_out^T + b_out (fp32 result)
// One wave per 32x32 output tile (2x2 WMMA register tiling).
__global__ __launch_bounds__(256) void out_proj_kernel(
    const _Float16* __restrict__ attn, const _Float16* __restrict__ woh,
    const float* __restrict__ b_out, float* __restrict__ out) {
  const int waveId = blockIdx.x * 8 + (threadIdx.x >> 5);
  const int nT = waveId & 31;          // 32 N-tiles of 32
  const int mT = waveId >> 5;          // 128 M-tiles of 32
  const int m0 = mT * 32, n0 = nT * 32;

  v8f c00 = vzero8(), c01 = vzero8(), c10 = vzero8(), c11 = vzero8();
#pragma unroll 2
  for (int k0 = 0; k0 < NC; k0 += 32) {
    v16h a0 = load_a_frag(attn, NC, m0, k0);
    v16h a1 = load_a_frag(attn, NC, m0 + 16, k0);
    v16h b0 = load_b_frag(woh, NC, n0, k0);
    v16h b1 = load_b_frag(woh, NC, n0 + 16, k0);
    c00 = wmma16(a0, b0, c00);
    c01 = wmma16(a0, b1, c01);
    c10 = wmma16(a1, b0, c10);
    c11 = wmma16(a1, b1, c11);
  }

  const int lane = threadIdx.x & 31;
  const int hi = lane >> 4;
  const int lo = lane & 15;
#pragma unroll
  for (int sub = 0; sub < 4; ++sub) {
    const int ms = m0 + (sub >> 1) * 16;
    const int ns = n0 + (sub & 1) * 16;
    const v8f acc = (sub == 0) ? c00 : (sub == 1) ? c01 : (sub == 2) ? c10 : c11;
    const int n = ns + lo;
    const float bias = b_out[n];
#pragma unroll
    for (int r = 0; r < 8; ++r) {
      int m = ms + r + hi * 8;
      out[((long)m << 10) + n] = acc[r] + bias;
    }
  }
}

// ---------------------------------------------------------------------------
extern "C" void kernel_launch(void* const* d_in, const int* in_sizes, int n_in,
                              void* d_out, int out_size, void* d_ws,
                              size_t ws_size, hipStream_t stream) {
  const float* x     = (const float*)d_in[0];
  const float* w_qkv = (const float*)d_in[1];
  const float* w_out = (const float*)d_in[2];
  const float* b_out = (const float*)d_in[3];
  float* out = (float*)d_out;

  char* ws = (char*)d_ws;
  _Float16* xh   = (_Float16*)(ws);               //  8 MB  x (f16)
  _Float16* wqh  = (_Float16*)(ws + 8388608);     //  6 MB  w_qkv (f16)
  _Float16* woh  = (_Float16*)(ws + 14680064);    //  2 MB  w_out (f16)
  _Float16* qf   = (_Float16*)(ws + 16777216);    //  8 MB  Q [b][h][t][d]
  _Float16* kf   = (_Float16*)(ws + 25165824);    //  8 MB  K [b][h][t][d]
  _Float16* vt   = (_Float16*)(ws + 33554432);    //  8 MB  V [b][h][d][t]
  _Float16* attn = (_Float16*)(ws + 41943040);    //  8 MB  attn out (f16)

  cvt_kernel<<<(NM * NC) / 256, 256, 0, stream>>>(x, xh, NM * NC);
  cvt_kernel<<<(N3C * NC) / 256, 256, 0, stream>>>(w_qkv, wqh, N3C * NC);
  cvt_kernel<<<(NC * NC) / 256, 256, 0, stream>>>(w_out, woh, NC * NC);

  // 128 M-tiles * 96 N-tiles (32x32 each) = 12288 wave-tiles / 8 waves
  qkv_rope_kernel<<<1536, 256, 0, stream>>>(xh, wqh, qf, kf, vt);
  // B*H*(T/16) = 4096 blocks
  attn_kernel<<<4096, 256, 0, stream>>>(qf, kf, vt, attn);
  // 128 M-tiles * 32 N-tiles (32x32 each) = 4096 wave-tiles / 8 waves
  out_proj_kernel<<<512, 256, 0, stream>>>(attn, woh, b_out, out);
}
